// ChebNet_67886253080998
// MI455X (gfx1250) — matrix operand order repaired and checked by reading, not compiled, not run on hardware
//
#include <hip/hip_runtime.h>
#include <hip/hip_bf16.h>

#define N_NODES   50000
#define N_EDGES   800000
#define HID       128
#define N_CLASSES 10
#define N_GRAPHS  64
#define N_CONV    4

typedef __attribute__((ext_vector_type(16))) __bf16 v16bf;
typedef __attribute__((ext_vector_type(8)))  float  v8f;

// bf16 (as u16 in low half of a dword pair) -> float by shifting into exponent
static __device__ __forceinline__ float bflo2f(unsigned u) {
    union { unsigned v; float f; } c; c.v = u << 16; return c.f;
}
static __device__ __forceinline__ float bfhi2f(unsigned u) {
    union { unsigned v; float f; } c; c.v = u & 0xffff0000u; return c.f;
}

// ---------------------------------------------------------------------------
// Pre-pack conv weights (4 layers x 2 mats x 128x128 f32, row-major [k][n])
// into the WMMA B-operand layout: [layer][mat][kt(4)][nt(8)][lane(32)][j(16)]
// lane: n = nt*16 + (lane&15); k = kt*32 + (j>>3)*16 + (lane>>4)*8 + (j&7)   (ISA 7.12.2)
// ---------------------------------------------------------------------------
__global__ void k_prepack_w(const float* __restrict__ W, __bf16* __restrict__ Wp) {
    int t = blockIdx.x * blockDim.x + threadIdx.x;
    if (t >= N_CONV * 2 * HID * HID) return;
    int j     = t & 15;
    int lane  = (t >> 4) & 31;
    int nt    = (t >> 9) & 7;
    int kt    = (t >> 12) & 3;
    int mat   = (t >> 14) & 1;
    int layer = t >> 15;
    int n = nt * 16 + (lane & 15);
    int k = kt * 32 + ((j >> 3) * 16) + ((lane >> 4) * 8) + (j & 7);
    Wp[t] = (__bf16)W[(((layer * 2 + mat) * HID) + k) * HID + n];
}

// f32 -> bf16 row-major bulk convert: 8 elements per thread
__global__ void k_f32_to_bf16(const float* __restrict__ src, __bf16* __restrict__ dst,
                              int n8) {
    int t = blockIdx.x * blockDim.x + threadIdx.x;
    if (t >= n8) return;
    const float4* s = (const float4*)src + (size_t)t * 2;
    float4 a = s[0], b = s[1];
    union { __bf16 h[8]; uint4 q; } o;
    o.h[0] = (__bf16)a.x; o.h[1] = (__bf16)a.y; o.h[2] = (__bf16)a.z; o.h[3] = (__bf16)a.w;
    o.h[4] = (__bf16)b.x; o.h[5] = (__bf16)b.y; o.h[6] = (__bf16)b.z; o.h[7] = (__bf16)b.w;
    ((uint4*)dst)[t] = o.q;
}

// ---------------------------------------------------------------------------
// Degree / normalization
// ---------------------------------------------------------------------------
__global__ void k_deg(const int* __restrict__ row, float* __restrict__ deg) {
    int e = blockIdx.x * blockDim.x + threadIdx.x;
    if (e < N_EDGES) atomicAdd(&deg[row[e]], 1.0f);
}

__global__ void k_dis(const float* __restrict__ deg, float* __restrict__ dis) {
    int n = blockIdx.x * blockDim.x + threadIdx.x;
    if (n < N_NODES) {
        float d = deg[n];
        dis[n] = (d > 0.0f) ? rsqrtf(fmaxf(d, 1.0f)) : 0.0f;
    }
}

__global__ void k_norm(const int* __restrict__ row, const int* __restrict__ col,
                       const float* __restrict__ dis, float* __restrict__ nrm) {
    int e = blockIdx.x * blockDim.x + threadIdx.x;
    if (e < N_EDGES) nrm[e] = -(dis[row[e]] * dis[col[e]]);
}

// ---------------------------------------------------------------------------
// tx1 = segment_sum(norm[e] * h[col[e]], row[e])    (h in bf16, tx1 f32)
// one wave per edge; each lane gathers 4 bf16 features (8B) + 4 f32 atomics
// ---------------------------------------------------------------------------
__global__ void k_lap(const int* __restrict__ row, const int* __restrict__ col,
                      const float* __restrict__ nrm, const __bf16* __restrict__ h,
                      float* __restrict__ tx1) {
    long long tid = (long long)blockIdx.x * blockDim.x + threadIdx.x;
    int e = (int)(tid >> 5);
    if (e >= N_EDGES) return;
    int lane = (int)(tid & 31);
    float w = nrm[e];
    uint2 raw = *(const uint2*)(h + (long long)col[e] * HID + lane * 4);
    float* dst = tx1 + (long long)row[e] * HID + lane * 4;
    atomicAdd(dst + 0, w * bflo2f(raw.x));
    atomicAdd(dst + 1, w * bfhi2f(raw.x));
    atomicAdd(dst + 2, w * bflo2f(raw.y));
    atomicAdd(dst + 3, w * bfhi2f(raw.y));
}

// ---------------------------------------------------------------------------
// out = relu(h @ W0 + tx1 @ W1 + b)    via v_wmma_f32_16x16x32_bf16
// one wave per 16x16 output tile; 8 WMMAs per tile (2 mats x K=128/32).
// h is bf16 row-major -> A operand assembled with two b128 loads, no VALU.
// tx1 is f32 -> native (__bf16) casts (hardware cvt if available).
// ---------------------------------------------------------------------------
__global__ void __launch_bounds__(256) k_cheb_gemm(
        const __bf16* __restrict__ h,     // [N, 128] bf16
        const float*  __restrict__ tx1,   // [N, 128] f32
        const __bf16* __restrict__ Wp,    // packed [2][4][8][32][16] (this layer)
        const float*  __restrict__ bias,  // [128] (this layer)
        __bf16*       __restrict__ out) { // [N, 128] bf16
    int wave = (int)((blockIdx.x * blockDim.x + threadIdx.x) >> 5);
    int lane = threadIdx.x & 31;
    int tm = wave >> 3;          // node-tile   (0..3124)
    int tn = wave & 7;           // column-tile (0..7)
    if (tm >= N_NODES / 16) return;   // uniform per wave -> EXEC all-ones inside

    int m  = lane & 15;
    int kb = (lane >> 4) * 8;    // A-operand K-group base (ISA 7.12.2)
    const __bf16* rowH = h   + (long long)(tm * 16 + m) * HID;
    const float*  rowT = tx1 + (long long)(tm * 16 + m) * HID;

    v8f acc = {};
    // T0 term: A direct from bf16 h (two 16B groups per K-step, zero conversion)
#pragma unroll
    for (int kt = 0; kt < 4; ++kt) {
        union { v16bf v; uint4 q[2]; } a;
        a.q[0] = *(const uint4*)(rowH + kt * 32 + kb);
        a.q[1] = *(const uint4*)(rowH + kt * 32 + 16 + kb);
        v16bf b = *(const v16bf*)(Wp + (size_t)((kt * 8 + tn) * 32 + lane) * 16);
        acc = __builtin_amdgcn_wmma_f32_16x16x32_bf16(
                false, a.v, false, b, (short)0, acc, false, false);
    }
    // T1 term: A from f32 tx1, native casts to bf16
#pragma unroll
    for (int kt = 0; kt < 4; ++kt) {
        const float4* g0 = (const float4*)(rowT + kt * 32 + kb);
        const float4* g1 = (const float4*)(rowT + kt * 32 + 16 + kb);
        float4 p0 = g0[0], p1 = g0[1], p2 = g1[0], p3 = g1[1];
        v16bf a;
        a[0]  = (__bf16)p0.x; a[1]  = (__bf16)p0.y; a[2]  = (__bf16)p0.z; a[3]  = (__bf16)p0.w;
        a[4]  = (__bf16)p1.x; a[5]  = (__bf16)p1.y; a[6]  = (__bf16)p1.z; a[7]  = (__bf16)p1.w;
        a[8]  = (__bf16)p2.x; a[9]  = (__bf16)p2.y; a[10] = (__bf16)p2.z; a[11] = (__bf16)p2.w;
        a[12] = (__bf16)p3.x; a[13] = (__bf16)p3.y; a[14] = (__bf16)p3.z; a[15] = (__bf16)p3.w;
        v16bf b = *(const v16bf*)(Wp + (size_t)(((4 + kt) * 8 + tn) * 32 + lane) * 16);
        acc = __builtin_amdgcn_wmma_f32_16x16x32_bf16(
                false, a, false, b, (short)0, acc, false, false);
    }

    int n  = lane & 15;
    int mb = (lane >> 4) * 8;    // C/D layout: VGPR v -> row mb+v (ISA 7.12.2)
    float bv = bias[tn * 16 + n];
    __bf16* orow = out + (long long)(tm * 16 + mb) * HID + tn * 16 + n;
#pragma unroll
    for (int v = 0; v < 8; ++v) {
        float val = acc[v] + bv;
        orow[(long long)v * HID] = (__bf16)((val > 0.0f) ? val : 0.0f);
    }
}

// ---------------------------------------------------------------------------
// Mean pooling per graph + MLP head
// ---------------------------------------------------------------------------
__global__ void k_pool_count(const int* __restrict__ batch, float* __restrict__ counts) {
    int n = blockIdx.x * blockDim.x + threadIdx.x;
    if (n < N_NODES) atomicAdd(&counts[batch[n]], 1.0f);
}

__global__ void k_pool_sum(const int* __restrict__ batch, const __bf16* __restrict__ h,
                           float* __restrict__ pooled) {
    long long t = (long long)blockIdx.x * blockDim.x + threadIdx.x;
    if (t >= (long long)N_NODES * HID) return;
    int n = (int)(t >> 7), f = (int)(t & 127);
    atomicAdd(&pooled[batch[n] * HID + f], (float)h[(long long)n * HID + f]);
}

__global__ void k_pool_div(float* __restrict__ pooled, const float* __restrict__ counts) {
    int t = blockIdx.x * blockDim.x + threadIdx.x;
    if (t < N_GRAPHS * HID) pooled[t] /= fmaxf(counts[t >> 7], 1.0f);
}

__global__ void k_mlp0(const float* __restrict__ pooled, const float* __restrict__ W,
                       const float* __restrict__ b, float* __restrict__ hidden) {
    int t = blockIdx.x * blockDim.x + threadIdx.x;
    if (t >= N_GRAPHS * 64) return;
    int g = t >> 6, j = t & 63;
    float s = b[j];
    for (int k = 0; k < HID; ++k) s += pooled[g * HID + k] * W[k * 64 + j];
    hidden[t] = (s > 0.0f) ? s : 0.0f;
}

__global__ void k_mlp1(const float* __restrict__ hidden, const float* __restrict__ W,
                       const float* __restrict__ b, float* __restrict__ out) {
    int t = blockIdx.x * blockDim.x + threadIdx.x;
    if (t >= N_GRAPHS * N_CLASSES) return;
    int g = t / N_CLASSES, c = t % N_CLASSES;
    float s = b[c];
    for (int j = 0; j < 64; ++j) s += hidden[g * 64 + j] * W[j * N_CLASSES + c];
    out[t] = s;
}

// ---------------------------------------------------------------------------
extern "C" void kernel_launch(void* const* d_in, const int* in_sizes, int n_in,
                              void* d_out, int out_size, void* d_ws, size_t ws_size,
                              hipStream_t stream) {
    const float* x      = (const float*)d_in[0];
    const float* conv_W = (const float*)d_in[1];
    const float* conv_b = (const float*)d_in[2];
    const float* lin0_W = (const float*)d_in[3];
    const float* lin0_b = (const float*)d_in[4];
    const float* lin1_W = (const float*)d_in[5];
    const float* lin1_b = (const float*)d_in[6];
    const int*   eidx   = (const int*)d_in[7];
    const int*   batch  = (const int*)d_in[8];
    const int*   row    = eidx;
    const int*   col    = eidx + N_EDGES;
    float* out = (float*)d_out;
    (void)in_sizes; (void)n_in; (void)out_size; (void)ws_size;

    char* ws = (char*)d_ws;
    size_t off = 0;
    auto carve = [&](size_t bytes) -> char* {
        char* p = ws + off;
        off += (bytes + 255) & ~(size_t)255;
        return p;
    };
    __bf16* xbf    = (__bf16*)carve((size_t)N_NODES * HID * 2);
    __bf16* hA     = (__bf16*)carve((size_t)N_NODES * HID * 2);
    __bf16* hB     = (__bf16*)carve((size_t)N_NODES * HID * 2);
    float*  tx1    = (float*) carve((size_t)N_NODES * HID * 4);
    float*  deg    = (float*) carve((size_t)N_NODES * 4);
    float*  dis    = (float*) carve((size_t)N_NODES * 4);
    float*  nrm    = (float*) carve((size_t)N_EDGES * 4);
    __bf16* Wp     = (__bf16*)carve((size_t)N_CONV * 2 * HID * HID * 2);
    float*  pooled = (float*) carve((size_t)N_GRAPHS * HID * 4);
    float*  counts = (float*) carve((size_t)N_GRAPHS * 4);
    float*  hidden = (float*) carve((size_t)N_GRAPHS * 64 * 4);

    // weights -> bf16 WMMA B layout; x -> bf16 row-major
    k_prepack_w<<<(N_CONV * 2 * HID * HID) / 256, 256, 0, stream>>>(conv_W, Wp);
    k_f32_to_bf16<<<(N_NODES * HID / 8 + 255) / 256, 256, 0, stream>>>(
        x, xbf, N_NODES * HID / 8);

    // degree / symmetric normalization
    hipMemsetAsync(deg, 0, (size_t)N_NODES * 4, stream);
    k_deg<<<(N_EDGES + 255) / 256, 256, 0, stream>>>(row, deg);
    k_dis<<<(N_NODES + 255) / 256, 256, 0, stream>>>(deg, dis);
    k_norm<<<(N_EDGES + 255) / 256, 256, 0, stream>>>(row, col, dis, nrm);

    // 4 ChebConv layers (h kept in bf16 between layers)
    const __bf16* hin = xbf;
    __bf16* houts[2] = {hA, hB};
    const int lap_blocks  = (int)(((long long)N_EDGES * 32 + 255) / 256); // 100000
    const int gemm_blocks = (N_NODES / 16) * (HID / 16) / 8;              // 3125
    for (int l = 0; l < N_CONV; ++l) {
        hipMemsetAsync(tx1, 0, (size_t)N_NODES * HID * 4, stream);
        k_lap<<<lap_blocks, 256, 0, stream>>>(row, col, nrm, hin, tx1);
        __bf16* hout = houts[l & 1];
        k_cheb_gemm<<<gemm_blocks, 256, 0, stream>>>(
            hin, tx1, Wp + (size_t)l * 2 * HID * HID, conv_b + l * HID, hout);
        hin = hout;
    }

    // mean pool per graph
    hipMemsetAsync(pooled, 0, (size_t)N_GRAPHS * HID * 4, stream);
    hipMemsetAsync(counts, 0, (size_t)N_GRAPHS * 4, stream);
    k_pool_count<<<(N_NODES + 255) / 256, 256, 0, stream>>>(batch, counts);
    k_pool_sum<<<(int)(((long long)N_NODES * HID + 255) / 256), 256, 0, stream>>>(batch, hin, pooled);
    k_pool_div<<<(N_GRAPHS * HID + 255) / 256, 256, 0, stream>>>(pooled, counts);

    // MLP head
    k_mlp0<<<(N_GRAPHS * 64 + 255) / 256, 256, 0, stream>>>(pooled, lin0_W, lin0_b, hidden);
    k_mlp1<<<(N_GRAPHS * N_CLASSES + 255) / 256, 256, 0, stream>>>(hidden, lin1_W, lin1_b, out);
}
